// StackRnn_7310034337937
// MI455X (gfx1250) — matrix operand order repaired and checked by reading, not compiled, not run on hardware
//
#include <hip/hip_runtime.h>
#include <stdint.h>

// ---------------- problem constants ----------------
#define B_   16
#define P_   32
#define S_   128
#define H_   256
#define N0_  512    // B*P sequences
#define G_   1024   // 4*H gates
#define O_   128

typedef __attribute__((ext_vector_type(16))) __bf16 v16bf;
typedef __attribute__((ext_vector_type(8)))  float  v8f;

// ---------------- helpers ----------------
__device__ __forceinline__ unsigned short f2bf(float x) {
    unsigned u = __float_as_uint(x);
    u += 0x7fffu + ((u >> 16) & 1u);          // round-to-nearest-even
    return (unsigned short)(u >> 16);
}
__device__ __forceinline__ float bf2f(unsigned short s) {
    return __uint_as_float(((unsigned)s) << 16);
}
__device__ __forceinline__ float sigm(float x) { return 1.0f / (1.0f + __expf(-x)); }

union FragU { v16bf v; uint4 q[2]; };

// A-matrix fragment (16x32 bf16, MxK). Lane holds row M = lane&15.
// K elements: e<8 -> K = e + koff ; e>=8 -> K = e + 8 + koff ; koff = (lane>=16)?8:0
__device__ __forceinline__ v16bf ldA(const unsigned short* rowPtr, int lane) {
    const int koff = (lane & 16) ? 8 : 0;
    FragU f;
    f.q[0] = *reinterpret_cast<const uint4*>(rowPtr + koff);
    f.q[1] = *reinterpret_cast<const uint4*>(rowPtr + koff + 16);
    return f.v;
}
// B-matrix fragment (32x16 bf16, KxN). Lane holds col N = lane&15.
// K elements: e<8 -> K = e + koff ; e>=8 -> K = e + 8 + koff ; koff = (lane>=16)?8:0. rowPtr points at W[n][k0] (row n of W = col n of B).
__device__ __forceinline__ v16bf ldB(const unsigned short* rowPtr, int lane) {
    const int koff = (lane & 16) ? 8 : 0;
    FragU f;
    f.q[0] = *reinterpret_cast<const uint4*>(rowPtr + koff);
    f.q[1] = *reinterpret_cast<const uint4*>(rowPtr + koff + 16);
    return f.v;
}

// ---------------- kernel 1: fp32 -> bf16 weight conversion ----------------
__global__ void cvt_bf16(const float* __restrict__ src, unsigned short* __restrict__ dst, int n) {
    int i = blockIdx.x * blockDim.x + threadIdx.x;
    if (i < n) dst[i] = f2bf(src[i]);
}

// ---------------- kernel 2: embedding gather -> bf16, time-major X[t][n][h] ----------------
__global__ void embed_kernel(const int* __restrict__ x, const float* __restrict__ emb,
                             unsigned short* __restrict__ X) {
    int row = blockIdx.x;            // row = t*512 + n
    int t = row >> 9;
    int n = row & 511;
    int tok = x[n * S_ + t];         // x is [B,P,S] == [n][s]
    int h = threadIdx.x;
    X[(size_t)row * H_ + h] = f2bf(emb[(size_t)tok * H_ + h]);
}

// ---------------- kernel 3: layer-0 bidirectional LSTM ----------------
// grid (32, 2): blockIdx.x = 16-sequence tile, blockIdx.y = direction (0 fwd, 1 bwd)
__global__ __launch_bounds__(256) void lstm0_kernel(
    const unsigned short* __restrict__ X,       // [S][512][256] bf16
    const int*            __restrict__ x_mask,  // [512][128] (1 = pad)
    const unsigned short* __restrict__ WihF, const unsigned short* __restrict__ WhhF,
    const float*          __restrict__ bF,
    const unsigned short* __restrict__ WihB, const unsigned short* __restrict__ WhhB,
    const float*          __restrict__ bB,
    unsigned short*       __restrict__ sen)     // [512][512] bf16 (fwd cols 0..255, bwd 256..511)
{
    __shared__ __align__(16) unsigned short gatesS[16][G_];     // bf16 gate preacts, 32 KB
    __shared__ __align__(16) unsigned short hbf[16][H_ + 8];    // bf16 h for WMMA A, padded rows

    const int dir  = blockIdx.y;
    const unsigned short* Wih  = dir ? WihB : WihF;
    const unsigned short* Whh  = dir ? WhhB : WhhF;
    const float*          bias = dir ? bB   : bF;

    const int n0   = blockIdx.x * 16;
    const int tid  = threadIdx.x;
    const int lane = tid & 31;
    const int wave = tid >> 5;
    const int em   = tid >> 4;            // elementwise: owned row (0..15)
    const int eh0  = (tid & 15) * 16;     // elementwise: owned hcol base

    float c_reg[16], h_reg[16];
    #pragma unroll
    for (int q = 0; q < 16; ++q) { c_reg[q] = 0.f; h_reg[q] = 0.f; }
    for (int i = tid; i < 16 * (H_ + 8); i += 256) ((unsigned short*)hbf)[i] = 0;
    __syncthreads();

    for (int step = 0; step < S_; ++step) {
        const int t = dir ? (S_ - 1 - step) : step;

        // ---- gate GEMM: gates[16,1024] = x_t @ Wih^T + h @ Whh^T + b ----
        v16bf Ax[8], Ah[8];
        const unsigned short* xrow = X + ((size_t)t * N0_ + n0 + (lane & 15)) * H_;
        #pragma unroll
        for (int kk = 0; kk < 8; ++kk) Ax[kk] = ldA(xrow + kk * 32, lane);
        const unsigned short* hrow = &hbf[lane & 15][0];
        #pragma unroll
        for (int kk = 0; kk < 8; ++kk) Ah[kk] = ldA(hrow + kk * 32, lane);

        for (int j = 0; j < 8; ++j) {                 // 8 N-tiles of 16 per wave
            const int ncol = wave * 128 + j * 16 + (lane & 15);
            const float bv = bias[ncol];
            v8f acc;
            #pragma unroll
            for (int v = 0; v < 8; ++v) acc[v] = bv;

            const unsigned short* wihRow = Wih + (size_t)ncol * H_;
            #pragma unroll
            for (int kk = 0; kk < 8; ++kk) {
                v16bf bfr = ldB(wihRow + kk * 32, lane);
                acc = __builtin_amdgcn_wmma_f32_16x16x32_bf16(false, Ax[kk], false, bfr,
                                                              (short)0, acc, false, false);
            }
            const unsigned short* whhRow = Whh + (size_t)ncol * H_;
            #pragma unroll
            for (int kk = 0; kk < 8; ++kk) {
                v16bf bfr = ldB(whhRow + kk * 32, lane);
                acc = __builtin_amdgcn_wmma_f32_16x16x32_bf16(false, Ah[kk], false, bfr,
                                                              (short)0, acc, false, false);
            }
            const int mb = (lane & 16) ? 8 : 0;       // D layout: VGPR v -> row v + mb
            #pragma unroll
            for (int v = 0; v < 8; ++v) gatesS[mb + v][ncol] = f2bf(acc[v]);
        }
        __syncthreads();

        // ---- elementwise LSTM cell update (c,h live in registers) ----
        const bool valid = (x_mask[(n0 + em) * S_ + t] == 0);
        #pragma unroll
        for (int q = 0; q < 16; ++q) {
            const int hc = eh0 + q;
            float gi = bf2f(gatesS[em][hc]);
            float gf = bf2f(gatesS[em][H_ + hc]);
            float gg = bf2f(gatesS[em][2 * H_ + hc]);
            float go = bf2f(gatesS[em][3 * H_ + hc]);
            float cn = sigm(gf) * c_reg[q] + sigm(gi) * tanhf(gg);
            float hn = sigm(go) * tanhf(cn);
            if (valid) { c_reg[q] = cn; h_reg[q] = hn; }
            hbf[em][hc] = f2bf(h_reg[q]);
        }
        __syncthreads();
    }

    // ---- write sentence representation (bf16, ready as layer-1 A operand) ----
    #pragma unroll
    for (int q = 0; q < 16; ++q) {
        const int hc = eh0 + q;
        sen[(size_t)(n0 + em) * (2 * H_) + dir * H_ + hc] = f2bf(h_reg[q]);
    }
}

// ---------------- kernel 4: layer-1 LSTM over paragraphs (N=16, K_in=512) ----------------
__global__ __launch_bounds__(256) void lstm1_kernel(
    const unsigned short* __restrict__ sen,    // [512][512] bf16, row = b*32 + p
    const unsigned short* __restrict__ Wih1,   // [1024][512] bf16
    const unsigned short* __restrict__ Whh1,   // [1024][256] bf16
    const float*          __restrict__ b1,
    float*                __restrict__ para)   // [16][256] fp32
{
    __shared__ __align__(16) unsigned short gatesS[16][G_];
    __shared__ __align__(16) unsigned short hbf[16][H_ + 8];

    const int tid  = threadIdx.x;
    const int lane = tid & 31;
    const int wave = tid >> 5;
    const int em   = tid >> 4;
    const int eh0  = (tid & 15) * 16;

    float c_reg[16], h_reg[16];
    #pragma unroll
    for (int q = 0; q < 16; ++q) { c_reg[q] = 0.f; h_reg[q] = 0.f; }
    for (int i = tid; i < 16 * (H_ + 8); i += 256) ((unsigned short*)hbf)[i] = 0;
    __syncthreads();

    for (int p = 0; p < P_; ++p) {
        v16bf Ah[8];
        const unsigned short* hrow = &hbf[lane & 15][0];
        #pragma unroll
        for (int kk = 0; kk < 8; ++kk) Ah[kk] = ldA(hrow + kk * 32, lane);
        const unsigned short* xrow = sen + (size_t)((lane & 15) * P_ + p) * (2 * H_);

        for (int j = 0; j < 8; ++j) {
            const int ncol = wave * 128 + j * 16 + (lane & 15);
            const float bv = b1[ncol];
            v8f acc;
            #pragma unroll
            for (int v = 0; v < 8; ++v) acc[v] = bv;

            const unsigned short* wr = Wih1 + (size_t)ncol * (2 * H_);
            #pragma unroll
            for (int kk = 0; kk < 16; ++kk) {     // K_in = 512
                v16bf a = ldA(xrow + kk * 32, lane);
                v16bf b = ldB(wr + kk * 32, lane);
                acc = __builtin_amdgcn_wmma_f32_16x16x32_bf16(false, a, false, b,
                                                              (short)0, acc, false, false);
            }
            const unsigned short* wh = Whh1 + (size_t)ncol * H_;
            #pragma unroll
            for (int kk = 0; kk < 8; ++kk) {      // K_h = 256
                v16bf b = ldB(wh + kk * 32, lane);
                acc = __builtin_amdgcn_wmma_f32_16x16x32_bf16(false, Ah[kk], false, b,
                                                              (short)0, acc, false, false);
            }
            const int mb = (lane & 16) ? 8 : 0;
            #pragma unroll
            for (int v = 0; v < 8; ++v) gatesS[mb + v][ncol] = f2bf(acc[v]);
        }
        __syncthreads();

        #pragma unroll
        for (int q = 0; q < 16; ++q) {
            const int hc = eh0 + q;
            float gi = bf2f(gatesS[em][hc]);
            float gf = bf2f(gatesS[em][H_ + hc]);
            float gg = bf2f(gatesS[em][2 * H_ + hc]);
            float go = bf2f(gatesS[em][3 * H_ + hc]);
            float cn = sigm(gf) * c_reg[q] + sigm(gi) * tanhf(gg);
            float hn = sigm(go) * tanhf(cn);
            c_reg[q] = cn; h_reg[q] = hn;     // all steps valid
            hbf[em][hc] = f2bf(hn);
        }
        __syncthreads();
    }

    #pragma unroll
    for (int q = 0; q < 16; ++q) para[em * H_ + eh0 + q] = h_reg[q];
}

// ---------------- kernel 5: final FC, out = para @ Wfc^T + bfc ----------------
__global__ void fc_kernel(const float* __restrict__ para, const float* __restrict__ Wfc,
                          const float* __restrict__ bfc, float* __restrict__ out) {
    int tid = blockIdx.x * blockDim.x + threadIdx.x;   // 2048 outputs
    int b = tid >> 7;
    int o = tid & 127;
    float s = bfc[o];
    const float* pr = para + b * H_;
    const float* wr = Wfc + o * H_;
    for (int h = 0; h < H_; ++h) s += pr[h] * wr[h];
    out[tid] = s;
}

// ---------------- host launcher ----------------
extern "C" void kernel_launch(void* const* d_in, const int* in_sizes, int n_in,
                              void* d_out, int out_size, void* d_ws, size_t ws_size,
                              hipStream_t stream) {
    const int*   x      = (const int*)  d_in[0];
    const int*   x_mask = (const int*)  d_in[1];
    const float* emb    = (const float*)d_in[2];
    const float* Wih0f  = (const float*)d_in[3];
    const float* Whh0f  = (const float*)d_in[4];
    const float* b0f    = (const float*)d_in[5];
    const float* Wih0b  = (const float*)d_in[6];
    const float* Whh0b  = (const float*)d_in[7];
    const float* b0b    = (const float*)d_in[8];
    const float* Wih1   = (const float*)d_in[9];
    const float* Whh1   = (const float*)d_in[10];
    const float* b1     = (const float*)d_in[11];
    const float* Wfc    = (const float*)d_in[12];
    const float* bfc    = (const float*)d_in[13];

    char* ws = (char*)d_ws;
    size_t off = 0;
    auto alloc = [&](size_t bytes) -> void* {
        off = (off + 255) & ~(size_t)255;
        void* p = ws + off;
        off += bytes;
        return p;
    };

    unsigned short* X     = (unsigned short*)alloc((size_t)S_ * N0_ * H_ * 2); // 32 MB
    unsigned short* WihFb = (unsigned short*)alloc((size_t)G_ * H_ * 2);
    unsigned short* WhhFb = (unsigned short*)alloc((size_t)G_ * H_ * 2);
    unsigned short* WihBb = (unsigned short*)alloc((size_t)G_ * H_ * 2);
    unsigned short* WhhBb = (unsigned short*)alloc((size_t)G_ * H_ * 2);
    unsigned short* Wih1b = (unsigned short*)alloc((size_t)G_ * 2 * H_ * 2);
    unsigned short* Whh1b = (unsigned short*)alloc((size_t)G_ * H_ * 2);
    unsigned short* sen   = (unsigned short*)alloc((size_t)N0_ * 2 * H_ * 2);
    float*          para  = (float*)         alloc((size_t)B_ * H_ * 4);

    const int nW  = G_ * H_;       // 262144
    const int nW1 = G_ * 2 * H_;   // 524288
    cvt_bf16<<<(nW  + 255) / 256, 256, 0, stream>>>(Wih0f, WihFb, nW);
    cvt_bf16<<<(nW  + 255) / 256, 256, 0, stream>>>(Whh0f, WhhFb, nW);
    cvt_bf16<<<(nW  + 255) / 256, 256, 0, stream>>>(Wih0b, WihBb, nW);
    cvt_bf16<<<(nW  + 255) / 256, 256, 0, stream>>>(Whh0b, WhhBb, nW);
    cvt_bf16<<<(nW1 + 255) / 256, 256, 0, stream>>>(Wih1,  Wih1b, nW1);
    cvt_bf16<<<(nW  + 255) / 256, 256, 0, stream>>>(Whh1,  Whh1b, nW);

    embed_kernel<<<S_ * N0_, 256, 0, stream>>>(x, emb, X);

    lstm0_kernel<<<dim3(32, 2), 256, 0, stream>>>(X, x_mask,
                                                  WihFb, WhhFb, b0f,
                                                  WihBb, WhhBb, b0b, sen);

    lstm1_kernel<<<1, 256, 0, stream>>>(sen, Wih1b, Whh1b, b1, para);

    fc_kernel<<<8, 256, 0, stream>>>(para, Wfc, bfc, (float*)d_out);
}
